// NaiveAttention_50955492000428
// MI455X (gfx1250) — matrix operand order repaired and checked by reading, not compiled, not run on hardware
//
#include <hip/hip_runtime.h>
#include <hip/hip_bf16.h>

// ---------------------------------------------------------------------------
// Naive attention (B=1, S=4096, D=512, H=8, Hd=64), fp32 in/out.
// MI455X (gfx1250, wave32) strategy:
//   * bf16 everywhere in the matmuls -> v_wmma_f32_16x16x32_bf16, fp32 acc
//   * flash-attention (online softmax), 512MB score matrix never materialized
//   * K/V tiles staged in LDS via the Tensor Data Mover (tensor_load_to_lds),
//     double-buffered, s_wait_tensorcnt + barrier; 8 waves share each tile
//   * V stored transposed [D][S] so every WMMA B-fragment read is contiguous
// ---------------------------------------------------------------------------

#define S_LEN 4096
#define DIM   512
#define NH    8
#define HD    64
#define ATT_SCALE 0.125f   // 1/sqrt(64)

typedef __attribute__((ext_vector_type(16))) __bf16 v16bf;
typedef __attribute__((ext_vector_type(8)))  __bf16 bf16x8;
typedef __attribute__((ext_vector_type(8)))  float  v8f;
typedef __attribute__((ext_vector_type(4)))  unsigned int v4u;
typedef __attribute__((ext_vector_type(8)))  int v8i;
typedef __attribute__((ext_vector_type(4)))  int v4i;

// ---- WMMA wrapper ----------------------------------------------------------
static __device__ __forceinline__ v8f wmma_bf16(v16bf a, v16bf b, v8f c) {
  return __builtin_amdgcn_wmma_f32_16x16x32_bf16(false, a, false, b,
                                                 (short)0, c, false, false);
}

// ---- Fragment loaders (CDNA5 ISA 16-bit A/B VGPR layouts) ------------------
static __device__ __forceinline__ v16bf load_frag_A(const __bf16* p, int ld,
                                                    int row0, int k0, int lane) {
  int row = row0 + (lane & 15);
  int kb  = k0 + ((lane >> 4) << 3);
  const __bf16* q = p + row * ld + kb;
  bf16x8 lo = *(const bf16x8*)(q);
  bf16x8 hi = *(const bf16x8*)(q + 16);
  v16bf a;
#pragma unroll
  for (int i = 0; i < 8; ++i) { a[i] = lo[i]; a[i + 8] = hi[i]; }
  return a;
}

static __device__ __forceinline__ v16bf load_frag_B(const __bf16* p, int ld,
                                                    int col0, int k0, int lane) {
  int col = col0 + (lane & 15);
  int kb  = k0 + ((lane >> 4) << 3);
  const __bf16* q = p + col * ld + kb;
  bf16x8 lo = *(const bf16x8*)(q);
  bf16x8 hi = *(const bf16x8*)(q + 16);
  v16bf b;
#pragma unroll
  for (int i = 0; i < 8; ++i) { b[i] = lo[i]; b[i + 8] = hi[i]; }
  return b;
}

// ---- TDM: 2-D bf16 tile (tile_d1 rows x tile_d0 elems) global -> LDS -------
// Descriptor packing per CDNA5 ISA.
// This toolchain exposes the 6-arg builtin:
//   (v4u g0, v8i g1, v4i g2, v4i g3, v8i g4, i32 cpol)
static __device__ __forceinline__ void tdm_load_tile_2d(
    unsigned lds_off, const __bf16* gptr,
    unsigned tensor_d0, unsigned tensor_d1, unsigned stride0,
    unsigned tile_d0, unsigned tile_d1) {
  unsigned long long ga = (unsigned long long)(size_t)gptr;
  v4u g0;
  g0[0] = 1u;                                            // count=1, user mode
  g0[1] = lds_off;                                       // LDS byte address
  g0[2] = (unsigned)(ga & 0xffffffffu);                  // global_addr[31:0]
  g0[3] = (unsigned)((ga >> 32) & 0x01ffffffu)           // global_addr[56:32]
        | (2u << 30);                                    // type = 2 ("image")
  v8i g1;
  g1[0] = (int)(1u << 16);                               // data_size = 2 bytes
  g1[1] = (int)((tensor_d0 & 0xffffu) << 16);            // tensor_dim0[15:0]
  g1[2] = (int)((tensor_d0 >> 16) |
                ((tensor_d1 & 0xffffu) << 16));          // dim0 hi | dim1 lo
  g1[3] = (int)((tensor_d1 >> 16) | (tile_d0 << 16));    // dim1 hi | tile_dim0
  g1[4] = (int)(tile_d1 & 0xffffu);                      // tile_dim1 (dim2=0)
  g1[5] = (int)stride0;                                  // tensor_dim0_stride
  g1[6] = 0;                                             // stride hi | dim1_stride
  g1[7] = 0;
  v4i z4 = {0, 0, 0, 0};
  v8i z8 = {0, 0, 0, 0, 0, 0, 0, 0};
  __builtin_amdgcn_tensor_load_to_lds(g0, g1, z4, z4, z8, 0);
}

// ---- fp32 -> bf16 elementwise convert --------------------------------------
__global__ void __launch_bounds__(256)
cvt_f32_bf16_kernel(const float* __restrict__ in, __bf16* __restrict__ out, int n) {
  int i = blockIdx.x * 256 + threadIdx.x;
  if (i < n) out[i] = (__bf16)in[i];
}

// ---- QKV projection: C = x @ W^T (M=4096,N=512,K=512), bf16 out ------------
__global__ void __launch_bounds__(256)
proj_qkv_kernel(const __bf16* __restrict__ xb,
                const __bf16* __restrict__ wq, const __bf16* __restrict__ wk,
                const __bf16* __restrict__ wv,
                __bf16* __restrict__ qb, __bf16* __restrict__ kbm,
                __bf16* __restrict__ vtb) {
  int lane = threadIdx.x & 31;
  int wave = threadIdx.x >> 5;
  int wm = wave >> 1, wn = wave & 1;
  int m0 = blockIdx.x * 128 + wm * 32;
  int n0 = blockIdx.y * 64 + wn * 32;
  int z  = blockIdx.z;
  const __bf16* W = (z == 0) ? wq : (z == 1) ? wk : wv;

  v8f acc[2][2] = {};
  for (int kk = 0; kk < DIM; kk += 32) {
    v16bf a0 = load_frag_A(xb, DIM, m0,      kk, lane);
    v16bf a1 = load_frag_A(xb, DIM, m0 + 16, kk, lane);
    v16bf b0 = load_frag_B(W, DIM, n0,      kk, lane);
    v16bf b1 = load_frag_B(W, DIM, n0 + 16, kk, lane);
    acc[0][0] = wmma_bf16(a0, b0, acc[0][0]);
    acc[0][1] = wmma_bf16(a0, b1, acc[0][1]);
    acc[1][0] = wmma_bf16(a1, b0, acc[1][0]);
    acc[1][1] = wmma_bf16(a1, b1, acc[1][1]);
  }
  int mo = 8 * (lane >> 4), nc = lane & 15;
#pragma unroll
  for (int i = 0; i < 2; ++i)
#pragma unroll
    for (int j = 0; j < 2; ++j)
#pragma unroll
      for (int r = 0; r < 8; ++r) {
        int m = m0 + i * 16 + mo + r;
        int n = n0 + j * 16 + nc;
        __bf16 val = (__bf16)acc[i][j][r];
        if (z == 0)      qb[m * DIM + n]  = val;
        else if (z == 1) kbm[m * DIM + n] = val;
        else             vtb[n * S_LEN + m] = val;   // V transposed
      }
}

// ---- Flash attention: 8 waves x 16-row Q tiles, TDM-staged K/V tiles -------
__global__ void __launch_bounds__(256)
flash_attn_kernel(const __bf16* __restrict__ qb, const __bf16* __restrict__ kbm,
                  const __bf16* __restrict__ vtb, __bf16* __restrict__ ao) {
  int lane = threadIdx.x & 31;
  int wave = threadIdx.x >> 5;
  int h  = blockIdx.y;
  int m0 = blockIdx.x * 128 + wave * 16;

  // double-buffered K tile [64 keys][64 dims], V tile [64 dims][64 keys]
  __shared__ __bf16 Kt[2][64 * 64];
  __shared__ __bf16 Vt[2][64 * 64];
  __shared__ __bf16 Pbuf[8 * 16 * 64];        // per-wave C->A layout bounce
  __bf16* pb = &Pbuf[wave * (16 * 64)];

  const int NT = S_LEN / 64;

  // prologue: wave 0 issues TDM loads for tile 0
  if (wave == 0) {
    tdm_load_tile_2d((unsigned)(size_t)&Kt[0][0], kbm + 0 * DIM + h * HD,
                     DIM, S_LEN, DIM, 64, 64);
    tdm_load_tile_2d((unsigned)(size_t)&Vt[0][0], vtb + (h * HD) * S_LEN + 0,
                     S_LEN, DIM, S_LEN, 64, 64);
  }

  v16bf qA0 = load_frag_A(qb, DIM, m0, h * HD + 0,  lane);
  v16bf qA1 = load_frag_A(qb, DIM, m0, h * HD + 32, lane);

  v8f O[4] = {};
  float mrow[8], lrow[8];
#pragma unroll
  for (int r = 0; r < 8; ++r) { mrow[r] = -3.0e38f; lrow[r] = 0.f; }

  for (int it = 0; it < NT; ++it) {
    int cur = it & 1;
    // previous iteration's readers are done (first iter: nothing to protect)
    __syncthreads();
    if (wave == 0) {
      if (it + 1 < NT) {
        int n1 = (it + 1) * 64;
        tdm_load_tile_2d((unsigned)(size_t)&Kt[cur ^ 1][0],
                         kbm + (size_t)n1 * DIM + h * HD, DIM, S_LEN, DIM, 64, 64);
        tdm_load_tile_2d((unsigned)(size_t)&Vt[cur ^ 1][0],
                         vtb + (size_t)(h * HD) * S_LEN + n1, S_LEN, DIM, S_LEN, 64, 64);
        __builtin_amdgcn_s_wait_tensorcnt(2);   // current tile's 2 loads done
      } else {
        __builtin_amdgcn_s_wait_tensorcnt(0);
      }
    }
    __syncthreads();                            // tile visible to all waves
    const __bf16* kt = &Kt[cur][0];
    const __bf16* vt = &Vt[cur][0];

    // S tile = Q (16xHd) * K^T (Hd x 64): B fragments from LDS
    v8f s[4];
#pragma unroll
    for (int j = 0; j < 4; ++j) {
      v16bf b0 = load_frag_B(kt, 64, j * 16, 0,  lane);
      v16bf b1 = load_frag_B(kt, 64, j * 16, 32, lane);
      v8f acc = {};
      acc = wmma_bf16(qA0, b0, acc);
      acc = wmma_bf16(qA1, b1, acc);
#pragma unroll
      for (int r = 0; r < 8; ++r) acc[r] *= ATT_SCALE;
      s[j] = acc;
    }
    // online softmax: row r+8*(lane/16) lives across the 16 lanes of a half
#pragma unroll
    for (int r = 0; r < 8; ++r) {
      float t = fmaxf(fmaxf(s[0][r], s[1][r]), fmaxf(s[2][r], s[3][r]));
      t = fmaxf(t, __shfl_xor(t, 1));
      t = fmaxf(t, __shfl_xor(t, 2));
      t = fmaxf(t, __shfl_xor(t, 4));
      t = fmaxf(t, __shfl_xor(t, 8));
      float nm = fmaxf(mrow[r], t);
      float alpha = __expf(mrow[r] - nm);
      mrow[r] = nm;
      float ps = 0.f;
#pragma unroll
      for (int j = 0; j < 4; ++j) {
        float pv = __expf(s[j][r] - nm);
        s[j][r] = pv;
        ps += pv;
      }
      ps += __shfl_xor(ps, 1);
      ps += __shfl_xor(ps, 2);
      ps += __shfl_xor(ps, 4);
      ps += __shfl_xor(ps, 8);
      lrow[r] = lrow[r] * alpha + ps;
#pragma unroll
      for (int j = 0; j < 4; ++j) O[j][r] *= alpha;
    }
    // C-layout -> A-layout via wave-private LDS bounce (in-order DS pipe)
    {
      int mloc = 8 * (lane >> 4);
      int cb = lane & 15;
#pragma unroll
      for (int j = 0; j < 4; ++j)
#pragma unroll
        for (int r = 0; r < 8; ++r)
          pb[(mloc + r) * 64 + j * 16 + cb] = (__bf16)s[j][r];
    }
    asm volatile("s_wait_dscnt 0" ::: "memory");
    v16bf pA0 = load_frag_A(pb, 64, 0, 0,  lane);
    v16bf pA1 = load_frag_A(pb, 64, 0, 32, lane);
    // O += P (16x64) * V (64xHd): B fragments from LDS V tile [dim][key]
#pragma unroll
    for (int j = 0; j < 4; ++j) {
      v16bf vb0 = load_frag_B(vt, 64, j * 16, 0,  lane);
      v16bf vb1 = load_frag_B(vt, 64, j * 16, 32, lane);
      O[j] = wmma_bf16(pA0, vb0, O[j]);
      O[j] = wmma_bf16(pA1, vb1, O[j]);
    }
  }
  // epilogue: normalize and store bf16 attention output [S][D]
  int mo = 8 * (lane >> 4), nc = lane & 15;
#pragma unroll
  for (int r = 0; r < 8; ++r) {
    float inv = 1.0f / lrow[r];
    int m = m0 + mo + r;
#pragma unroll
    for (int j = 0; j < 4; ++j) {
      int c = h * HD + j * 16 + nc;
      ao[m * DIM + c] = (__bf16)(O[j][r] * inv);
    }
  }
}

// ---- Output projection: out = attn_out @ Wo^T, fp32 result -----------------
__global__ void __launch_bounds__(256)
proj_out_kernel(const __bf16* __restrict__ ab, const __bf16* __restrict__ wo,
                float* __restrict__ out) {
  int lane = threadIdx.x & 31;
  int wave = threadIdx.x >> 5;
  int wm = wave >> 1, wn = wave & 1;
  int m0 = blockIdx.x * 128 + wm * 32;
  int n0 = blockIdx.y * 64 + wn * 32;

  v8f acc[2][2] = {};
  for (int kk = 0; kk < DIM; kk += 32) {
    v16bf a0 = load_frag_A(ab, DIM, m0,      kk, lane);
    v16bf a1 = load_frag_A(ab, DIM, m0 + 16, kk, lane);
    v16bf b0 = load_frag_B(wo, DIM, n0,      kk, lane);
    v16bf b1 = load_frag_B(wo, DIM, n0 + 16, kk, lane);
    acc[0][0] = wmma_bf16(a0, b0, acc[0][0]);
    acc[0][1] = wmma_bf16(a0, b1, acc[0][1]);
    acc[1][0] = wmma_bf16(a1, b0, acc[1][0]);
    acc[1][1] = wmma_bf16(a1, b1, acc[1][1]);
  }
  int mo = 8 * (lane >> 4), nc = lane & 15;
#pragma unroll
  for (int i = 0; i < 2; ++i)
#pragma unroll
    for (int j = 0; j < 2; ++j)
#pragma unroll
      for (int r = 0; r < 8; ++r) {
        int m = m0 + i * 16 + mo + r;
        int n = n0 + j * 16 + nc;
        out[m * DIM + n] = acc[i][j][r];
      }
}

// ---------------------------------------------------------------------------
extern "C" void kernel_launch(void* const* d_in, const int* in_sizes, int n_in,
                              void* d_out, int out_size, void* d_ws, size_t ws_size,
                              hipStream_t stream) {
  const float* x  = (const float*)d_in[0];
  const float* Wq = (const float*)d_in[1];
  const float* Wk = (const float*)d_in[2];
  const float* Wv = (const float*)d_in[3];
  const float* Wo = (const float*)d_in[4];
  float* out = (float*)d_out;

  const int NX = S_LEN * DIM;     // 2,097,152
  const int NW = DIM * DIM;       //   262,144

  __bf16* xb  = (__bf16*)d_ws;
  __bf16* wqb = xb  + NX;
  __bf16* wkb = wqb + NW;
  __bf16* wvb = wkb + NW;
  __bf16* wob = wvb + NW;
  __bf16* qb  = wob + NW;
  __bf16* kbm = qb  + NX;
  __bf16* vtb = kbm + NX;   // [D][S] transposed V
  __bf16* aob = vtb + NX;   // attention output, bf16 [S][D]
  // total ~= 23 MB of workspace

  cvt_f32_bf16_kernel<<<(NX + 255) / 256, 256, 0, stream>>>(x,  xb,  NX);
  cvt_f32_bf16_kernel<<<(NW + 255) / 256, 256, 0, stream>>>(Wq, wqb, NW);
  cvt_f32_bf16_kernel<<<(NW + 255) / 256, 256, 0, stream>>>(Wk, wkb, NW);
  cvt_f32_bf16_kernel<<<(NW + 255) / 256, 256, 0, stream>>>(Wv, wvb, NW);
  cvt_f32_bf16_kernel<<<(NW + 255) / 256, 256, 0, stream>>>(Wo, wob, NW);

  proj_qkv_kernel<<<dim3(S_LEN / 128, DIM / 64, 3), 256, 0, stream>>>(
      xb, wqb, wkb, wvb, qb, kbm, vtb);

  flash_attn_kernel<<<dim3(S_LEN / 128, NH), 256, 0, stream>>>(qb, kbm, vtb, aob);

  proj_out_kernel<<<dim3(S_LEN / 128, DIM / 64), 256, 0, stream>>>(aob, wob, out);
}
